// NPSN_54915451847272
// MI455X (gfx1250) — matrix operand order repaired and checked by reading, not compiled, not run on hardware
//
#include <hip/hip_runtime.h>

typedef __attribute__((ext_vector_type(16))) __bf16 v16bf;
typedef __attribute__((ext_vector_type(8)))  float  v8f;
typedef __attribute__((ext_vector_type(2)))  float  v2f;

#if defined(__has_builtin)
#  if __has_builtin(__builtin_amdgcn_wmma_f32_16x16x4_f32)
#    define HAVE_WMMA_F32X4 1
#  endif
#endif
#ifndef HAVE_WMMA_F32X4
#  define HAVE_WMMA_F32X4 0
#endif

union BF16x16 {
  v16bf v;
  unsigned short u[16];
  unsigned int   pack[8];
  uint4          q[2];
};

__device__ __forceinline__ unsigned short f2bf(float f) {
  unsigned int x = __float_as_uint(f);
  x += 0x7fffu + ((x >> 16) & 1u);          // round-to-nearest-even
  return (unsigned short)(x >> 16);
}

// ---------------------------------------------------------------------------
#define BB   64
#define NN   512
#define HH   4
#define INF_ 16
#define OF_  64
#define H1_  1024
#define H2_  512
#define OD_  40

// ===========================================================================
// Kernel 1: per-head projection via V_WMMA_F32_16X16X4_F32 (f32 precision)
//   h_prime stored TRANSPOSED: hp_t[B,H,64(o),512(n)] bf16 (K-major for WMMA B)
//   + skip = sum_h h_prime (f32), + attn_src/attn_dst scores
// grid = B*(N/16), 128 threads (wave h = head h)
// ===========================================================================
__global__ void k_gat_proj(const float* __restrict__ x, const float* __restrict__ w,
                           const float* __restrict__ a_src, const float* __restrict__ a_dst,
                           unsigned short* __restrict__ hp_t, float* __restrict__ skip,
                           float* __restrict__ srcv, float* __restrict__ dstv)
{
  const int b  = blockIdx.x >> 5;
  const int nt = blockIdx.x & 31;
  const int tid  = threadIdx.x;
  const int h    = tid >> 5;
  const int lane = tid & 31;
  const int mrow = lane & 15;
  const int hi   = lane >> 4;

  __shared__ float lds[HH][16][OF_];   // 16 KB

  const float* node = x + ((size_t)b * NN + nt * 16) * INF_;

  v8f c[4] = {};

#if HAVE_WMMA_F32X4
  for (int k0 = 0; k0 < INF_; k0 += 4) {
    v2f a;
    a.x = node[mrow * INF_ + k0 + 2 * hi];
    a.y = node[mrow * INF_ + k0 + 2 * hi + 1];
    #pragma unroll
    for (int f = 0; f < 4; ++f) {
      const int o = f * 16 + mrow;
      v2f bb;
      bb.x = w[(h * INF_ + k0 + 2 * hi) * OF_ + o];
      bb.y = w[(h * INF_ + k0 + 2 * hi + 1) * OF_ + o];
      c[f] = __builtin_amdgcn_wmma_f32_16x16x4_f32(false, a, false, bb,
                                                   (short)0, c[f], false, false);
    }
  }
#else
  for (int f = 0; f < 4; ++f)
    for (int r = 0; r < 8; ++r) {
      const int M = r + 8 * hi, o = f * 16 + mrow;
      float acc = 0.f;
      for (int k = 0; k < INF_; ++k)
        acc += node[M * INF_ + k] * w[(h * INF_ + k) * OF_ + o];
      c[f][r] = acc;
    }
#endif

  #pragma unroll
  for (int f = 0; f < 4; ++f)
    #pragma unroll
    for (int r = 0; r < 8; ++r)
      lds[h][r + 8 * hi][f * 16 + mrow] = c[f][r];
  __syncthreads();

  // skip-sum over heads (f32)
  for (int i = tid; i < 16 * OF_; i += 128) {
    const int r = i >> 6, o = i & 63;
    skip[((size_t)b * NN + nt * 16 + r) * OF_ + o] =
        lds[0][r][o] + lds[1][r][o] + lds[2][r][o] + lds[3][r][o];
  }
  // h_prime transposed bf16: hp_t[((b*H+h)*64 + o)*512 + n]
  for (int i = tid; i < HH * OF_ * 16; i += 128) {
    const int hh = i >> 10, rem = i & 1023, o = rem >> 4, r = rem & 15;
    hp_t[(((size_t)b * HH + hh) * OF_ + o) * NN + nt * 16 + r] = f2bf(lds[hh][r][o]);
  }
  // attention scores: 128 tasks = 4 heads x 16 rows x {src,dst}
  {
    const int hh = tid >> 5, r = (tid >> 1) & 15, isdst = tid & 1;
    const float* av = (isdst ? a_dst : a_src) + hh * OF_;
    float acc = 0.f;
    for (int o = 0; o < OF_; ++o) acc += lds[hh][r][o] * av[o];
    float* dp = isdst ? dstv : srcv;
    dp[((size_t)b * HH + hh) * NN + nt * 16 + r] = acc;
  }
}

// ===========================================================================
// Kernel 2: per-row softmax statistics over leaky(s_n * d_m)
// ===========================================================================
__global__ void k_softmax_stats(const float* __restrict__ srcv, const float* __restrict__ dstv,
                                float* __restrict__ rmax, float* __restrict__ rsum)
{
  const int wave = threadIdx.x >> 5, lane = threadIdx.x & 31;
  const size_t row = (size_t)blockIdx.x * 8 + wave;  // ((b*H+h)*N + n)
  const size_t bh  = row >> 9;
  const float  s   = srcv[row];
  const float* d   = dstv + (bh << 9);

  float e[16];
  float mx = -3.4e38f;
  #pragma unroll
  for (int i = 0; i < 16; ++i) {
    float v = s * d[lane + 32 * i];
    v = (v < 0.f) ? 0.2f * v : v;
    e[i] = v;
    mx = fmaxf(mx, v);
  }
  for (int off = 16; off; off >>= 1) mx = fmaxf(mx, __shfl_xor(mx, off, 32));
  float sum = 0.f;
  #pragma unroll
  for (int i = 0; i < 16; ++i) sum += __expf(e[i] - mx);
  for (int off = 16; off; off >>= 1) sum += __shfl_xor(sum, off, 32);
  if (lane == 0) { rmax[row] = mx; rsum[row] = sum; }
}

// ===========================================================================
// Kernel 3: attention GEMM, A-tile (softmax*mask probabilities) built in regs,
// h_prime staged from the K-major transposed layout -> B frags = 2x ds_read_b128
// grid = B*(N/16), 128 threads (wave h = head h)
// ===========================================================================
__global__ void k_attn_gemm(const unsigned short* __restrict__ hp_t, const float* __restrict__ mask,
                            const float* __restrict__ srcv, const float* __restrict__ dstv,
                            const float* __restrict__ rmax, const float* __restrict__ rsum,
                            const float* __restrict__ skip, const float* __restrict__ bias,
                            unsigned short* __restrict__ A1)
{
  const int b  = blockIdx.x >> 5;
  const int nt = blockIdx.x & 31;
  const int tid = threadIdx.x, h = tid >> 5, lane = tid & 31;
  const int mrow = lane & 15, hi = lane >> 4;

  __shared__ __align__(16) unsigned short stageT[HH][OF_][32]; // [o][k], 16 KB
  __shared__ float red[HH][4][32][8];                          // 16 KB

  const int    row = nt * 16 + mrow;
  const size_t bh  = (size_t)b * HH + h;
  const float  s   = srcv[(bh << 9) + row];
  const float  mx  = rmax[(bh << 9) + row];
  const float  inv = 1.0f / rsum[(bh << 9) + row];
  const float* dv  = dstv + (bh << 9);
  const float* mrw = mask + ((size_t)b * NN + row) * NN;
  const unsigned short* hpb = hp_t + bh * OF_ * NN;   // [o][n], row stride 512

  v8f c[4] = {};

  for (int mc = 0; mc < NN; mc += 32) {
    // stage hp_t[o=0..64, m=mc..mc+32): 64 rows x 64B, uint4 global->LDS
    uint4* dst4 = reinterpret_cast<uint4*>(&stageT[h][0][0]);
    for (int idx = lane; idx < 256; idx += 32) {
      const int o = idx >> 2, part = idx & 3;
      dst4[idx] = *(const uint4*)(hpb + (size_t)o * NN + mc + part * 8);
    }
    // prefetch next chunk while this one is consumed
    if (mc + 32 < NN) {
      __builtin_prefetch(hpb + (size_t)(2 * lane) * NN + mc + 32, 0, 2);
      __builtin_prefetch(hpb + (size_t)(2 * lane + 1) * NN + mc + 32, 0, 2);
    }
    __syncthreads();

    // A fragment: 16x32 bf16 attention probabilities, built in registers
    BF16x16 aP;
    #pragma unroll
    for (int t = 0; t < 8; ++t) {
      const int kk = (t < 4) ? (8 * hi + 2 * t) : (16 + 8 * hi + 2 * (t - 4));
      #pragma unroll
      for (int j = 0; j < 2; ++j) {
        const int m = mc + kk + j;
        float e = s * dv[m];
        e = (e < 0.f) ? 0.2f * e : e;
        const float p = __expf(e - mx) * inv * mrw[m];
        aP.u[2 * t + j] = f2bf(p);
      }
    }

    // B fragments: contiguous packed pairs -> 2x b128 LDS reads each
    #pragma unroll
    for (int f = 0; f < 4; ++f) {
      BF16x16 bF;
      const int o = f * 16 + mrow;
      bF.q[0] = *(const uint4*)&stageT[h][o][8 * hi];
      bF.q[1] = *(const uint4*)&stageT[h][o][16 + 8 * hi];
      c[f] = __builtin_amdgcn_wmma_f32_16x16x32_bf16(false, aP.v, false, bF.v,
                                                     (short)0, c[f], false, false);
    }
    __syncthreads();
  }

  // head reduction via LDS, then epilogue: + skip + bias, ReLU, bf16
  #pragma unroll
  for (int f = 0; f < 4; ++f)
    #pragma unroll
    for (int r = 0; r < 8; ++r) red[h][f][lane][r] = c[f][r];
  __syncthreads();

  for (int i = tid; i < 16 * OF_; i += 128) {
    const int r = i >> 6, o = i & 63;
    const int f = o >> 4, ln = (o & 15) + ((r & 8) ? 16 : 0), vg = r & 7;
    const float acc = red[0][f][ln][vg] + red[1][f][ln][vg] +
                      red[2][f][ln][vg] + red[3][f][ln][vg];
    const size_t idx = ((size_t)b * NN + nt * 16 + r) * OF_ + o;
    const float v = acc + skip[idx] + bias[o];
    A1[idx] = f2bf(fmaxf(v, 0.f));
  }
}

// ===========================================================================
// Kernel 4: bf16 WMMA GEMM, 128x64 block tile, 4 waves x (32 rows x 64 cols).
//   C = A[MxK] * BT[N][K]^T + bias; EPI 0: ReLU->bf16, EPI 1: sigmoid+clip->f32
// Weights pre-transposed (K-major) so both A and B frags are 2x ds_read_b128.
// ===========================================================================
template <int EPI>
__global__ void k_gemm(const unsigned short* __restrict__ A, const unsigned short* __restrict__ BT,
                       const float* __restrict__ bias, void* __restrict__ out,
                       int M, int N, int K, int realN, int ldo)
{
  const int nb   = N >> 6;
  const int brow = blockIdx.x / nb, bcol = blockIdx.x % nb;
  const int tid = threadIdx.x, w = tid >> 5, lane = tid & 31;
  const int mrow = lane & 15, hi = lane >> 4;

  __shared__ __align__(16) unsigned short sA[128][32];   // 8 KB, [row][k]
  __shared__ __align__(16) unsigned short sBT[64][32];   // 4 KB, [n][k]

  const unsigned short* Ablk  = A  + (size_t)(brow * 128) * K;
  const unsigned short* BTblk = BT + (size_t)(bcol * 64) * K;

  v8f c[2][4] = {};

  for (int kc = 0; kc < K; kc += 32) {
    uint4* sa4 = reinterpret_cast<uint4*>(&sA[0][0]);
    for (int idx = tid; idx < 512; idx += 128) {         // A: 128 rows x 64B
      const int r = idx >> 2, part = idx & 3;
      sa4[idx] = *(const uint4*)(Ablk + (size_t)r * K + kc + part * 8);
    }
    uint4* sb4 = reinterpret_cast<uint4*>(&sBT[0][0]);
    for (int idx = tid; idx < 256; idx += 128) {         // B: 64 rows x 64B
      const int n = idx >> 2, part = idx & 3;
      sb4[idx] = *(const uint4*)(BTblk + (size_t)n * K + kc + part * 8);
    }
    if (kc + 32 < K) {                                   // prefetch next chunk
      __builtin_prefetch(Ablk + (size_t)tid * K + kc + 32, 0, 2);
      if (tid < 64) __builtin_prefetch(BTblk + (size_t)tid * K + kc + 32, 0, 2);
    }
    __syncthreads();

    BF16x16 aF[2];
    #pragma unroll
    for (int rf = 0; rf < 2; ++rf) {
      const int r = w * 32 + rf * 16 + mrow;
      aF[rf].q[0] = *(const uint4*)&sA[r][8 * hi];
      aF[rf].q[1] = *(const uint4*)&sA[r][16 + 8 * hi];
    }
    #pragma unroll
    for (int f = 0; f < 4; ++f) {
      BF16x16 bF;
      const int n = f * 16 + mrow;
      bF.q[0] = *(const uint4*)&sBT[n][8 * hi];
      bF.q[1] = *(const uint4*)&sBT[n][16 + 8 * hi];
      #pragma unroll
      for (int rf = 0; rf < 2; ++rf)
        c[rf][f] = __builtin_amdgcn_wmma_f32_16x16x32_bf16(false, aF[rf].v, false, bF.v,
                                                           (short)0, c[rf][f], false, false);
    }
    __syncthreads();
  }

  #pragma unroll
  for (int rf = 0; rf < 2; ++rf)
    #pragma unroll
    for (int f = 0; f < 4; ++f) {
      const int col = bcol * 64 + f * 16 + mrow;
      #pragma unroll
      for (int r = 0; r < 8; ++r) {
        const int row = brow * 128 + w * 32 + rf * 16 + r + 8 * hi;
        if (col < realN) {
          const float v = c[rf][f][r] + bias[col];
          if (EPI == 0) {
            ((unsigned short*)out)[(size_t)row * ldo + col] = f2bf(fmaxf(v, 0.f));
          } else {
            float sg = 1.f / (1.f + __expf(-v));
            sg = fminf(fmaxf(sg, 0.01f), 0.99f);
            ((float*)out)[(size_t)row * ldo + col] = sg;
          }
        }
      }
    }
}

// ===========================================================================
// Kernel 5: f32 [K][N] -> bf16 transposed [Npad][K], zero-padded rows
// ===========================================================================
__global__ void k_cvt_t(const float* __restrict__ src, unsigned short* __restrict__ dst,
                        int K, int N, int Npad)
{
  const int total = Npad * K;
  for (int i = blockIdx.x * 256 + threadIdx.x; i < total; i += gridDim.x * 256) {
    const int n = i / K, k = i - n * K;
    dst[i] = (n < N) ? f2bf(src[(size_t)k * N + n]) : (unsigned short)0;
  }
}

// ===========================================================================
extern "C" void kernel_launch(void* const* d_in, const int* in_sizes, int n_in,
                              void* d_out, int out_size, void* d_ws, size_t ws_size,
                              hipStream_t stream)
{
  const float* x     = (const float*)d_in[0];
  const float* mask  = (const float*)d_in[1];
  const float* w     = (const float*)d_in[2];
  const float* a_src = (const float*)d_in[3];
  const float* a_dst = (const float*)d_in[4];
  const float* bias  = (const float*)d_in[5];
  const float* W1    = (const float*)d_in[6];
  const float* b1    = (const float*)d_in[7];
  const float* W2    = (const float*)d_in[8];
  const float* b2    = (const float*)d_in[9];
  const float* W3    = (const float*)d_in[10];
  const float* b3    = (const float*)d_in[11];
  float* out = (float*)d_out;

  const size_t BN = (size_t)BB * NN, BHN = BN * HH;

  char* p = (char*)d_ws;
  auto carve = [&](size_t bytes) { char* q = p; p += (bytes + 255) & ~(size_t)255; return q; };
  unsigned short* hp_t = (unsigned short*)carve(BHN * OF_ * 2); // 16.8 MB, [B,H,64,512]
  float* srcv = (float*)carve(BHN * 4);
  float* dstv = (float*)carve(BHN * 4);
  float* rmax = (float*)carve(BHN * 4);
  float* rsum = (float*)carve(BHN * 4);
  float* skip = (float*)carve(BN * OF_ * 4);                    // 8.4 MB
  unsigned short* A1   = (unsigned short*)carve(BN * OF_ * 2);
  unsigned short* A2   = (unsigned short*)carve(BN * H1_ * 2);  // 67 MB
  unsigned short* A3   = (unsigned short*)carve(BN * H2_ * 2);  // 33.5 MB
  unsigned short* WT1  = (unsigned short*)carve((size_t)H1_ * OF_ * 2);  // [1024][64]
  unsigned short* WT2  = (unsigned short*)carve((size_t)H2_ * H1_ * 2);  // [512][1024]
  unsigned short* WT3  = (unsigned short*)carve((size_t)64 * H2_ * 2);   // [64][512]

  // transposed bf16 weights (W3 zero-padded to 64 output cols)
  k_cvt_t<<<256, 256, 0, stream>>>(W1, WT1, OF_, H1_, H1_);
  k_cvt_t<<<1024, 256, 0, stream>>>(W2, WT2, H1_, H2_, H2_);
  k_cvt_t<<<64, 256, 0, stream>>>(W3, WT3, H2_, OD_, 64);

  // GAT projection + scores (f32 WMMA)
  k_gat_proj<<<BB * (NN / 16), 128, 0, stream>>>(x, w, a_src, a_dst, hp_t, skip, srcv, dstv);

  // softmax statistics
  k_softmax_stats<<<(BB * HH * NN) / 8, 256, 0, stream>>>(srcv, dstv, rmax, rsum);

  // attention GEMM fused with skip/bias/ReLU -> A1 (bf16)
  k_attn_gemm<<<BB * (NN / 16), 128, 0, stream>>>(hp_t, mask, srcv, dstv, rmax, rsum,
                                                  skip, bias, A1);

  // MLP (128x64 tiles)
  const int Mrows = (int)BN;                       // 32768
  k_gemm<0><<<(Mrows / 128) * (H1_ / 64), 128, 0, stream>>>(A1, WT1, b1, A2,
                                                            Mrows, H1_, OF_, H1_, H1_);
  k_gemm<0><<<(Mrows / 128) * (H2_ / 64), 128, 0, stream>>>(A2, WT2, b2, A3,
                                                            Mrows, H2_, H1_, H2_, H2_);
  k_gemm<1><<<(Mrows / 128) * 1, 128, 0, stream>>>(A3, WT3, b3, out,
                                                   Mrows, 64, H2_, OD_, OD_);
  (void)in_sizes; (void)n_in; (void)out_size; (void)ws_size;
}
